// ThinkingLSTM_7524782702943
// MI455X (gfx1250) — matrix operand order repaired and checked by reading, not compiled, not run on hardware
//
#include <hip/hip_runtime.h>
#include <hip/hip_bf16.h>

// ---------------- problem constants ----------------
#define I_SZ        64
#define H_SZ        128
#define G_SZ        512           // 4*H
#define V_SZ        64
#define NUM_CLASSES 63
#define MAX_THINK   16
#define B_SZ        512
#define S_SZ        128
#define N_DEC       125           // S - 3
#define BT          16            // batch rows per workgroup
#define WG_THREADS  256           // 8 waves of 32

// ---------------- types ----------------
typedef __attribute__((ext_vector_type(16))) __bf16        v16bf;
typedef __attribute__((ext_vector_type(8)))  float         v8f;
typedef __attribute__((ext_vector_type(4)))  unsigned int  v4u;
typedef __attribute__((ext_vector_type(8)))  int           v8i_t;
typedef __attribute__((ext_vector_type(4)))  int           v4i_t;

union FragU { v16bf bf; unsigned int u[8]; };
struct BFragRaw { v4u lo, hi; };                 // 32B, 16B-aligned pieces
union BFragCvt { BFragRaw raw; v16bf bf; };

// ---------------- weight fragment layout in workspace / LDS ----------------
// One WMMA B-fragment (32x16 bf16) = 1024B, per-lane 32B contiguous.
#define FRAG_BYTES 1024
#define WIH_FRAGS  64      // 32 gate tiles * 2 K-chunks (K=64)
#define WHH_FRAGS  128     // 32 gate tiles * 4 K-chunks (K=128)
#define WFC_FRAGS  16      // 4 out tiles  * 4 K-chunks
#define WS_WIH_OFF 0
#define WS_WHH_OFF (WIH_FRAGS * FRAG_BYTES)                  // 65536
#define WS_WFC_OFF (WS_WHH_OFF + WHH_FRAGS * FRAG_BYTES)     // 196608
#define WS_WEIGHTS_BYTES (WS_WFC_OFF + WFC_FRAGS * FRAG_BYTES) // 212992

// ---------------- LDS layout (dynamic shared) ----------------
#define LDS_WIH  0u
#define LDS_WHH  65536u
#define LDS_WFC  196608u
#define LDS_H    212992u                 // 16x128 bf16 (ushort) = 4096
#define LDS_X    (LDS_H + 4096u)         // 16x64  bf16          = 2048
#define LDS_C    (LDS_X + 2048u)         // 16x128 f32           = 8192
#define LDS_G    (LDS_C + 8192u)         // 16x512 f32           = 32768
#define LDS_L    (LDS_G + 32768u)        // 16x64  f32           = 4096
#define LDS_DONE (LDS_L + 4096u)         // 16 ints
#define LDS_TOTAL (LDS_DONE + 64u)       // ~258 KB < 320 KB/WGP

#if defined(__has_builtin)
#if __has_builtin(__builtin_amdgcn_tensor_load_to_lds) && __has_builtin(__builtin_amdgcn_s_wait_tensorcnt)
#define HAVE_TDM 1
#endif
#if __has_builtin(__builtin_amdgcn_tanhf)
#define FAST_TANH(x) __builtin_amdgcn_tanhf(x)      // V_TANH_F32 (TRANS32)
#endif
#endif
#ifndef FAST_TANH
#define FAST_TANH(x) tanhf(x)
#endif

// ---------------- small helpers ----------------
__device__ __forceinline__ unsigned short f32_to_bf16(float f) {
  union { float f; unsigned u; } x; x.f = f;
  unsigned r = x.u + 0x7FFFu + ((x.u >> 16) & 1u);   // RNE
  return (unsigned short)(r >> 16);
}

__device__ __forceinline__ float sigmoidf_(float x) {
  return 1.0f / (1.0f + __expf(-x));
}

__device__ __forceinline__ float gumbel_hash(unsigned a, unsigned b, unsigned c) {
  unsigned h = (a + 0x9E3779B1u) * 0x85EBCA77u;
  h ^= (b + 0x7F4A7C15u) * 0xC2B2AE3Du;
  h ^= (c + 0x165667B1u) * 0x27D4EB2Fu;
  h ^= h >> 15; h *= 0x2C1B3C6Du; h ^= h >> 12; h *= 0x297A2D39u; h ^= h >> 15;
  float u = ((float)(h >> 8) + 0.5f) * (1.0f / 16777216.0f);
  return -__logf(-__logf(u));
}

__device__ __forceinline__ v8f wmma_bf16(v16bf a, v16bf b, v8f c) {
  // (neg_a, A, neg_b, B, c_mod, C, reuse_a, reuse_b)
  return __builtin_amdgcn_wmma_f32_16x16x32_bf16(false, a, false, b, (short)0, c, false, false);
}

// B fragment: pre-packed, per-lane 32B contiguous + 32B aligned in LDS.
// Two 16B vector loads -> ds_load_b128 pair.
__device__ __forceinline__ v16bf load_b_frag(const unsigned char* smem, unsigned off, int lane) {
  const v4u* p = (const v4u*)(smem + off + (unsigned)lane * 32u);
  BFragCvt c;
  c.raw.lo = p[0];
  c.raw.hi = p[1];
  return c.bf;
}

// A fragment (16x32 bf16) gathered from a row-major bf16 LDS tile.
// ISA layout: lanes 0-15 M=0..15 {V0..3:K=0..7, V4..7:K=16..23}; lanes 16-31 +8.
__device__ __forceinline__ v16bf load_a_frag(const unsigned char* smem, unsigned off,
                                             int stride, int lane, int kbase) {
  const unsigned short* h = (const unsigned short*)(smem + off);
  int m = lane & 15, hi = lane >> 4;
  FragU f;
#pragma unroll
  for (int v = 0; v < 8; ++v) {
    int kk = kbase + ((v & 3) * 2) + ((v >> 2) * 16) + hi * 8;
    f.u[v] = *(const unsigned int*)(h + m * stride + kk);   // bf16 pair (K even)
  }
  return f.bf;
}

// ---------------- pack kernel: f32 weights -> bf16 WMMA B-fragments ----------------
__global__ __launch_bounds__(256) void pack_weights(const float* __restrict__ Wih,
                                                    const float* __restrict__ Whh,
                                                    const float* __restrict__ Wfc,
                                                    unsigned int* __restrict__ ws) {
  int f = blockIdx.x;
  int tid = threadIdx.x;
  int lane = tid >> 3;   // 0..31
  int v    = tid & 7;    // 0..7
  const float* W; int in_dim, tile, kc; unsigned base;
  if (f < WIH_FRAGS) {
    W = Wih; in_dim = I_SZ; tile = f >> 1; kc = f & 1;
    base = WS_WIH_OFF / 4 + (unsigned)f * 256u;
  } else if (f < WIH_FRAGS + WHH_FRAGS) {
    int g = f - WIH_FRAGS;
    W = Whh; in_dim = H_SZ; tile = g >> 2; kc = g & 3;
    base = WS_WHH_OFF / 4 + (unsigned)g * 256u;
  } else {
    int g = f - WIH_FRAGS - WHH_FRAGS;
    W = Wfc; in_dim = H_SZ; tile = g >> 2; kc = g & 3;
    base = WS_WFC_OFF / 4 + (unsigned)g * 256u;
  }
  int n = lane & 15, khalf = lane >> 4;
  int col = tile * 16 + n;                 // output (gate / class) column
  int K   = kc * 32 + (v & 3) * 2 + (v >> 2) * 16 + khalf * 8;  // B element (K, n) = W[col][K]
  unsigned short lo = f32_to_bf16(W[col * in_dim + K]);
  unsigned short hi = f32_to_bf16(W[col * in_dim + K + 1]);
  ws[base + (unsigned)lane * 8u + (unsigned)v] = ((unsigned)hi << 16) | (unsigned)lo;
}

// ---------------- per-step building blocks ----------------
__device__ __forceinline__ void compute_gates(unsigned char* smem, int wave, int lane,
                                              const float* bias4) {
  v16bf ax0 = load_a_frag(smem, LDS_X, I_SZ, lane, 0);
  v16bf ax1 = load_a_frag(smem, LDS_X, I_SZ, lane, 32);
  v16bf ah0 = load_a_frag(smem, LDS_H, H_SZ, lane, 0);
  v16bf ah1 = load_a_frag(smem, LDS_H, H_SZ, lane, 32);
  v16bf ah2 = load_a_frag(smem, LDS_H, H_SZ, lane, 64);
  v16bf ah3 = load_a_frag(smem, LDS_H, H_SZ, lane, 96);
  float* gates = (float*)(smem + LDS_G);
  int nl = lane & 15, mb = (lane >> 4) * 8;
#pragma unroll
  for (int j = 0; j < 4; ++j) {
    int tile = wave * 4 + j;
    v8f acc;
#pragma unroll
    for (int r = 0; r < 8; ++r) acc[r] = bias4[j];
    acc = wmma_bf16(ax0, load_b_frag(smem, LDS_WIH + (unsigned)(tile * 2 + 0) * FRAG_BYTES, lane), acc);
    acc = wmma_bf16(ax1, load_b_frag(smem, LDS_WIH + (unsigned)(tile * 2 + 1) * FRAG_BYTES, lane), acc);
    acc = wmma_bf16(ah0, load_b_frag(smem, LDS_WHH + (unsigned)(tile * 4 + 0) * FRAG_BYTES, lane), acc);
    acc = wmma_bf16(ah1, load_b_frag(smem, LDS_WHH + (unsigned)(tile * 4 + 1) * FRAG_BYTES, lane), acc);
    acc = wmma_bf16(ah2, load_b_frag(smem, LDS_WHH + (unsigned)(tile * 4 + 2) * FRAG_BYTES, lane), acc);
    acc = wmma_bf16(ah3, load_b_frag(smem, LDS_WHH + (unsigned)(tile * 4 + 3) * FRAG_BYTES, lane), acc);
#pragma unroll
    for (int r = 0; r < 8; ++r)
      gates[(mb + r) * G_SZ + tile * 16 + nl] = acc[r];
  }
}

__device__ __forceinline__ void compute_logits(unsigned char* smem, int wave, int lane,
                                               float biasfc) {
  if (wave >= 4) return;                       // wave-uniform branch; EXEC full inside
  v16bf a0 = load_a_frag(smem, LDS_H, H_SZ, lane, 0);
  v16bf a1 = load_a_frag(smem, LDS_H, H_SZ, lane, 32);
  v16bf a2 = load_a_frag(smem, LDS_H, H_SZ, lane, 64);
  v16bf a3 = load_a_frag(smem, LDS_H, H_SZ, lane, 96);
  v8f acc;
#pragma unroll
  for (int r = 0; r < 8; ++r) acc[r] = biasfc;
  acc = wmma_bf16(a0, load_b_frag(smem, LDS_WFC + (unsigned)(wave * 4 + 0) * FRAG_BYTES, lane), acc);
  acc = wmma_bf16(a1, load_b_frag(smem, LDS_WFC + (unsigned)(wave * 4 + 1) * FRAG_BYTES, lane), acc);
  acc = wmma_bf16(a2, load_b_frag(smem, LDS_WFC + (unsigned)(wave * 4 + 2) * FRAG_BYTES, lane), acc);
  acc = wmma_bf16(a3, load_b_frag(smem, LDS_WFC + (unsigned)(wave * 4 + 3) * FRAG_BYTES, lane), acc);
  float* L = (float*)(smem + LDS_L);
  int nl = lane & 15, mb = (lane >> 4) * 8;
#pragma unroll
  for (int r = 0; r < 8; ++r)
    L[(mb + r) * V_SZ + wave * 16 + nl] = acc[r];
}

__device__ __forceinline__ void update_hc(unsigned char* smem, int tid, bool use_done) {
  int m  = tid >> 4;           // batch row 0..15
  int j0 = (tid & 15) * 8;     // hidden cols, 8 each
  const float* gates = (const float*)(smem + LDS_G);
  float* c = (float*)(smem + LDS_C);
  unsigned short* h = (unsigned short*)(smem + LDS_H);
  const int* done = (const int*)(smem + LDS_DONE);
  if (use_done && done[m]) return;
#pragma unroll
  for (int q = 0; q < 8; ++q) {
    int j = j0 + q;
    float gi = gates[m * G_SZ + j];
    float gf = gates[m * G_SZ + H_SZ + j];
    float gg = gates[m * G_SZ + 2 * H_SZ + j];
    float go = gates[m * G_SZ + 3 * H_SZ + j];
    float cn = sigmoidf_(gf) * c[m * H_SZ + j] + sigmoidf_(gi) * FAST_TANH(gg);
    c[m * H_SZ + j] = cn;
    h[m * H_SZ + j] = f32_to_bf16(sigmoidf_(go) * FAST_TANH(cn));
  }
}

__device__ __forceinline__ void sample_row(unsigned char* smem, int m, int b0,
                                           unsigned phase, int step, bool think) {
  const float* L = (const float*)(smem + LDS_L);
  int* done = (int*)(smem + LDS_DONE);
  unsigned short* X = (unsigned short*)(smem + LDS_X);
  if (think && done[m]) return;        // done rows freeze h/c/inp
  float best = -3.0e38f; int bi = 0;
  for (int v = 0; v < V_SZ; ++v) {
    float val = L[m * V_SZ + v] +
                gumbel_hash(phase, (unsigned)(step * B_SZ + b0 + m), (unsigned)v);
    if (val > best) { best = val; bi = v; }
  }
  for (int v = 0; v < V_SZ; ++v)
    X[m * V_SZ + v] = (v == bi) ? (unsigned short)0x3F80 : (unsigned short)0;
  if (think && bi == NUM_CLASSES) done[m] = 1;
}

__device__ __forceinline__ void load_x_tile(unsigned char* smem, const float* __restrict__ x,
                                            int b0, int t, int tid) {
  int m = tid >> 4, k0 = (tid & 15) * 4;
  float4 xv = *(const float4*)(x + ((size_t)(b0 + m) * S_SZ + t) * I_SZ + k0);
  unsigned short* X = (unsigned short*)(smem + LDS_X);
  X[m * I_SZ + k0 + 0] = f32_to_bf16(xv.x);
  X[m * I_SZ + k0 + 1] = f32_to_bf16(xv.y);
  X[m * I_SZ + k0 + 2] = f32_to_bf16(xv.z);
  X[m * I_SZ + k0 + 3] = f32_to_bf16(xv.w);
}

// ---------------- persistent LSTM kernel ----------------
__global__ __launch_bounds__(WG_THREADS)
void lstm_persistent(const float* __restrict__ x,
                     const float* __restrict__ b_ih, const float* __restrict__ b_hh,
                     const float* __restrict__ b_fc,
                     const unsigned int* __restrict__ wfrag,
                     float* __restrict__ out) {
  extern __shared__ unsigned char smem[];
  const int tid  = threadIdx.x;
  const int wave = tid >> 5;
  const int lane = tid & 31;
  const int b0   = blockIdx.x * BT;

  // ---- stage 208KB of pre-packed bf16 weight fragments into LDS (TDM) ----
#ifdef HAVE_TDM
  if (wave == 0) {
    unsigned long long ga = (unsigned long long)wfrag;   // uniform (kernarg)
    const unsigned rows = WS_WEIGHTS_BYTES / 256u;       // 832 rows x 64 dwords
    v4u g0;
    g0[0] = 1u;                                          // count=1, user descriptor
    g0[1] = 0u;                                          // lds_addr = 0 (dyn LDS base)
    g0[2] = (unsigned)(ga & 0xFFFFFFFFull);              // global_addr[31:0]
    g0[3] = (unsigned)((ga >> 32) & 0x1FFFFFFull) | (2u << 30);   // addr[56:32] | type=2
    v8i_t g1;
    g1[0] = (int)(2u << 16);                             // data_size = 4B dwords
    g1[1] = (int)(64u << 16);                            // tensor_dim0[15:0]
    g1[2] = (int)((rows & 0xFFFFu) << 16);               // tdim0 hi=0 | tensor_dim1 lo
    g1[3] = (int)(((rows >> 16) & 0xFFFFu) | (64u << 16)); // tdim1 hi | tile_dim0=64
    g1[4] = (int)(rows & 0xFFFFu);                       // tile_dim1
    g1[5] = 64;                                          // tensor_dim0_stride[31:0]
    g1[6] = 0;
    g1[7] = 0;
    v4i_t gz;  gz[0] = 0; gz[1] = 0; gz[2] = 0; gz[3] = 0;
    v8i_t gz8; gz8[0] = 0; gz8[1] = 0; gz8[2] = 0; gz8[3] = 0;
    gz8[4] = 0; gz8[5] = 0; gz8[6] = 0; gz8[7] = 0;
    // 6-arg variant (clang-23 / therock headers): (g0, g1, g2, g3, g4, cpol)
    __builtin_amdgcn_tensor_load_to_lds(g0, g1, gz, gz, gz8, 0);
    __builtin_amdgcn_s_wait_tensorcnt((short)0);
  }
#else
  {
    unsigned int* dst = (unsigned int*)smem;
    for (int i = tid; i < WS_WEIGHTS_BYTES / 4; i += WG_THREADS) dst[i] = wfrag[i];
  }
#endif

  // ---- init h=0, c=0, done=0 ----
  {
    float* c = (float*)(smem + LDS_C);
    unsigned short* h = (unsigned short*)(smem + LDS_H);
    for (int i = tid; i < BT * H_SZ; i += WG_THREADS) { c[i] = 0.0f; h[i] = 0; }
    if (tid < BT) ((int*)(smem + LDS_DONE))[tid] = 0;
  }

  // ---- per-lane bias preload (C layout: value depends only on N = lane%16) ----
  float bias4[4];
#pragma unroll
  for (int j = 0; j < 4; ++j) {
    int col = (wave * 4 + j) * 16 + (lane & 15);
    bias4[j] = b_ih[col] + b_hh[col];
  }
  float biasfc = (wave < 4) ? b_fc[wave * 16 + (lane & 15)] : 0.0f;
  __syncthreads();

  // ===== encoder: 128 steps over x =====
  for (int t = 0; t < S_SZ; ++t) {
    load_x_tile(smem, x, b0, t, tid);
    if (t + 1 < S_SZ) {
      const float* nxt = x + ((size_t)(b0 + (tid >> 4)) * S_SZ + (t + 1)) * I_SZ;
      __builtin_prefetch(nxt, 0, 0);                     // global_prefetch_b8
    }
    __syncthreads();
    compute_gates(smem, wave, lane, bias4);
    __syncthreads();
    update_hc(smem, tid, false);
    __syncthreads();
  }
  // LDS_X now holds x[:, S-1, :] == initial `inp` for the think phase.

  // ===== think: up to MAX_THINK+1 gumbel-hard steps with done-masking =====
  for (int ts = 0; ts <= MAX_THINK; ++ts) {
    compute_gates(smem, wave, lane, bias4);
    __syncthreads();
    update_hc(smem, tid, true);
    __syncthreads();
    compute_logits(smem, wave, lane, biasfc);
    __syncthreads();
    if (tid < BT) sample_row(smem, tid, b0, 0u, ts, true);
    __syncthreads();
  }

  // ===== decode: 125 steps, emit logits[:, :63] =====
  for (int td = 0; td < N_DEC; ++td) {
    compute_gates(smem, wave, lane, bias4);
    __syncthreads();
    update_hc(smem, tid, false);
    __syncthreads();
    compute_logits(smem, wave, lane, biasfc);
    __syncthreads();
    const float* L = (const float*)(smem + LDS_L);
    for (int e = tid; e < BT * NUM_CLASSES; e += WG_THREADS) {
      int m = e / NUM_CLASSES, v = e % NUM_CLASSES;
      out[((size_t)(b0 + m) * N_DEC + td) * NUM_CLASSES + v] = L[m * V_SZ + v];
    }
    if (tid < BT) sample_row(smem, tid, b0, 1u, td, false);
    __syncthreads();
  }
}

// ---------------- launch ----------------
extern "C" void kernel_launch(void* const* d_in, const int* in_sizes, int n_in,
                              void* d_out, int out_size, void* d_ws, size_t ws_size,
                              hipStream_t stream) {
  (void)in_sizes; (void)n_in; (void)out_size; (void)ws_size;
  const float* x   = (const float*)d_in[0];
  const float* Wih = (const float*)d_in[1];
  const float* Whh = (const float*)d_in[2];
  const float* bih = (const float*)d_in[3];
  const float* bhh = (const float*)d_in[4];
  const float* Wfc = (const float*)d_in[5];
  const float* bfc = (const float*)d_in[6];
  unsigned int* wfrag = (unsigned int*)d_ws;

  (void)hipFuncSetAttribute((const void*)lstm_persistent,
                            hipFuncAttributeMaxDynamicSharedMemorySize, (int)LDS_TOTAL);

  pack_weights<<<WIH_FRAGS + WHH_FRAGS + WFC_FRAGS, 256, 0, stream>>>(Wih, Whh, Wfc, wfrag);
  lstm_persistent<<<B_SZ / BT, WG_THREADS, LDS_TOTAL, stream>>>(x, bih, bhh, bfc, wfrag,
                                                                (float*)d_out);
}
